// CustomSelfAttentionWithBias_60155311948109
// MI455X (gfx1250) — matrix operand order repaired and checked by reading, not compiled, not run on hardware
//
#include <hip/hip_runtime.h>
#include <hip/hip_bf16.h>

// ---------------------------------------------------------------------------
// CDNA5 (gfx1250) self-attention with column-0 emphasis.
// B=2, T=2048, C=1024, H=16, hd=64.
// Matmuls: v_wmma_f32_16x16x32_bf16.
// GEMM tile staging: TDM tensor_load_to_lds (TENSORcnt).
// Attention tile staging: global_load_async_to_lds_b128 (ASYNCcnt).
// ---------------------------------------------------------------------------

typedef __bf16 bf16_t;
typedef __attribute__((ext_vector_type(16))) __bf16 v16bf;
typedef __attribute__((ext_vector_type(8)))  __bf16 v8bf;
typedef __attribute__((ext_vector_type(8)))  float  v8f;
typedef __attribute__((ext_vector_type(4)))  unsigned int v4u;
typedef __attribute__((ext_vector_type(8)))  unsigned int v8u;
typedef __attribute__((address_space(3))) bf16_t lds_bf16_t;

#define T_SEQ 2048
#define HD 64
#define NHEAD 16
#define CDIM 1024
#define BROWS 4096            // B*T
#define NEG_BIG (-3.0e38f)

static __device__ __forceinline__ v16bf join8(v8bf lo, v8bf hi) {
  return __builtin_shufflevector(lo, hi, 0,1,2,3,4,5,6,7,8,9,10,11,12,13,14,15);
}

// ---- CDNA5 async global->LDS copy (ASYNCcnt-tracked, no VGPR data) --------
static __device__ __forceinline__ void async_b128(const bf16_t* g, bf16_t* l) {
  unsigned int loff = (unsigned int)(size_t)(lds_bf16_t*)l;    // LDS byte offset
  unsigned long long ga = (unsigned long long)(size_t)g;
  asm volatile("global_load_async_to_lds_b128 %0, %1, off"
               :: "v"(loff), "v"(ga) : "memory");
}
static __device__ __forceinline__ void async_wait0() {
  asm volatile("s_wait_asynccnt 0x0" ::: "memory");
}

// ---- CDNA5 Tensor Data Mover: 2-D bf16 tile, global -> LDS ----------------
// D# per CDNA5 ISA. 2-group form (VADDR2/3 = NULL).
// tile: tile_d1 rows x tile_d0 elems (2B each), row stride = stride0 elems.
// TDM packs rows back-to-back in LDS starting at lds_off.
static __device__ __forceinline__ void tdm_load_2d(unsigned int lds_off,
                                                   const bf16_t* gtile,
                                                   unsigned int tensor_d0,
                                                   unsigned int tensor_d1,
                                                   unsigned int stride0,
                                                   unsigned int tile_d0,
                                                   unsigned int tile_d1) {
  unsigned long long ga = (unsigned long long)(size_t)gtile;
  v4u g0;
  g0[0] = 1u;                                          // count=1, user mode
  g0[1] = lds_off;                                     // lds_addr
  g0[2] = (unsigned int)(ga & 0xffffffffull);          // global_addr[31:0]
  g0[3] = (unsigned int)((ga >> 32) & 0x01ffffffull)   // global_addr[56:32]
          | (2u << 30);                                // type = 2 ("image")
  v8u g1;
  g1[0] = (1u << 16);                                  // data_size=1 (2 bytes)
  g1[1] = (tensor_d0 & 0xffffu) << 16;                 // tensor_dim0[15:0]
  g1[2] = (tensor_d0 >> 16) | ((tensor_d1 & 0xffffu) << 16);
  g1[3] = (tensor_d1 >> 16) | ((tile_d0 & 0xffffu) << 16);   // tile_dim0
  g1[4] = (tile_d1 & 0xffffu);                         // tile_dim1 (tile_dim2=0)
  g1[5] = stride0;                                     // tensor_dim0_stride[31:0]
  g1[6] = 0u;                                          // stride0[47:32], s1[15:0]
  g1[7] = 0u;                                          // tensor_dim1_stride hi
  asm volatile("tensor_load_to_lds %0, %1" :: "s"(g0), "s"(g1) : "memory");
}
static __device__ __forceinline__ void tensor_wait0() {
  __builtin_amdgcn_s_wait_tensorcnt((short)0);
}

// ---------------------------------------------------------------------------
// f32 -> bf16 bulk convert (x), 4 elements/thread.
// ---------------------------------------------------------------------------
__global__ __launch_bounds__(256)
void cvt_f32_to_bf16(const float* __restrict__ in, bf16_t* __restrict__ out, int n) {
  int i = (blockIdx.x * 256 + threadIdx.x) * 4;
  if (i + 3 < n) {
    float4 v = *(const float4*)(in + i);
    out[i + 0] = (bf16_t)v.x;
    out[i + 1] = (bf16_t)v.y;
    out[i + 2] = (bf16_t)v.z;
    out[i + 3] = (bf16_t)v.w;
  }
}

// ---------------------------------------------------------------------------
// W[K,N] f32 -> Wt[N,K] bf16 (32x32 LDS tiles, both sides coalesced).
// ---------------------------------------------------------------------------
__global__ __launch_bounds__(256)
void transpose_cvt(const float* __restrict__ W, bf16_t* __restrict__ Wt,
                   int K, int N) {
  __shared__ float tile[32][33];
  const int n0 = blockIdx.x * 32;
  const int k0 = blockIdx.y * 32;
  const int tx = threadIdx.x & 31;
  const int ty = threadIdx.x >> 5;          // 0..7
  #pragma unroll
  for (int i = 0; i < 4; ++i) {
    const int r = ty + i * 8;               // k within tile
    tile[r][tx] = W[(size_t)(k0 + r) * N + n0 + tx];
  }
  __syncthreads();
  #pragma unroll
  for (int i = 0; i < 4; ++i) {
    const int r = ty + i * 8;               // n within tile
    Wt[(size_t)(n0 + r) * K + k0 + tx] = (bf16_t)tile[tx][r];
  }
}

// ---------------------------------------------------------------------------
// GEMM: C[M,N] = A[M,K] (bf16 rm) * Bt[N,K] (bf16 rm, i.e. B transposed).
// 128x128 tile / 256 threads, k-step 32.  Both LDS tiles staged by the TDM:
// wave 0 issues one tensor_load_to_lds per tile, waits TENSORcnt, barrier.
// Wave grid 4(M) x 2(N): each wave owns 32x64 (2x4 fragments).
// ---------------------------------------------------------------------------
template <typename OutT>
__global__ __launch_bounds__(256)
void gemm_bf16_wmma(const bf16_t* __restrict__ A, const bf16_t* __restrict__ Bt,
                    OutT* __restrict__ C, int M, int N, int K) {
  __shared__ __align__(32) bf16_t As[128][32];   // As[m][k]
  __shared__ __align__(32) bf16_t Bs[128][32];   // Bs[n][k]

  const int bm = blockIdx.y * 128;
  const int bn = blockIdx.x * 128;
  const int tid  = threadIdx.x;
  const int lane = tid & 31;
  const int wave = tid >> 5;
  const int l16  = lane & 15;
  const int hi   = lane >> 4;
  const int wm   = (wave >> 1) * 32;
  const int wn   = (wave & 1) * 64;

  const unsigned int asOff = (unsigned int)(size_t)(lds_bf16_t*)&As[0][0];
  const unsigned int bsOff = (unsigned int)(size_t)(lds_bf16_t*)&Bs[0][0];
  const bf16_t* atile = A  + (size_t)bm * K;
  const bf16_t* btile = Bt + (size_t)bn * K;

  v8f acc[2][4] = {};

  for (int k0 = 0; k0 < K; k0 += 32) {
    if (wave == 0) {
      // 128 rows x 32 bf16, row stride K elems -> packed LDS rows of 64 B
      tdm_load_2d(asOff, atile + k0, (unsigned)K, 128u, (unsigned)K, 32u, 128u);
      tdm_load_2d(bsOff, btile + k0, (unsigned)K, 128u, (unsigned)K, 32u, 128u);
      tensor_wait0();
    }
    __syncthreads();

    #pragma unroll
    for (int i = 0; i < 2; ++i) {
      const int ar = wm + 16 * i + l16;
      v8bf alo = *(const v8bf*)&As[ar][hi * 16];
      v8bf ahi = *(const v8bf*)&As[ar][hi * 16 + 8];
      v16bf af = join8(alo, ahi);
      #pragma unroll
      for (int j = 0; j < 4; ++j) {
        v16bf bfrag = *(const v16bf*)&Bs[wn + 16 * j + l16][hi * 16];
        acc[i][j] = __builtin_amdgcn_wmma_f32_16x16x32_bf16(
            false, af, false, bfrag, (short)0, acc[i][j], false, false);
      }
    }
    __syncthreads();
  }

  #pragma unroll
  for (int i = 0; i < 2; ++i) {
    #pragma unroll
    for (int j = 0; j < 4; ++j) {
      const int col = bn + wn + 16 * j + l16;
      #pragma unroll
      for (int v = 0; v < 8; ++v) {
        const int row = bm + wm + 16 * i + hi * 8 + v;
        C[(size_t)row * N + col] = (OutT)acc[i][j][v];
      }
    }
  }
}

// ---------------------------------------------------------------------------
// Flash attention: block = (64-query tile, head, batch); 128 threads (4 waves).
// qkv bf16 [B*T, 3C]; out bf16 [B*T, C] (columns h*64+d).
// ---------------------------------------------------------------------------
__global__ __launch_bounds__(128)
void attn_flash_wmma(const bf16_t* __restrict__ qkv, bf16_t* __restrict__ out) {
  __shared__ __align__(32) bf16_t qs[64][64];   // Q tile [qrow][d]
  __shared__ __align__(32) bf16_t ks[32][64];   // K tile [key][d] (Bt layout)
  __shared__ __align__(32) bf16_t vt[64][32];   // V tile transposed [d][key]
  __shared__ __align__(32) float  sS[64][32];   // raw scores
  __shared__ __align__(32) bf16_t pb[64][32];   // probabilities
  __shared__ float mrow[64], lrow[64], arow[64];

  const int qt = blockIdx.x;
  const int h  = blockIdx.y;
  const int b  = blockIdx.z;
  const int q0 = qt * 64;
  const int tid  = threadIdx.x;
  const int lane = tid & 31;
  const int wave = tid >> 5;            // rows 16*wave .. +15
  const int l16  = lane & 15;
  const int hi   = lane >> 4;

  const size_t rstride = 3 * CDIM;
  const bf16_t* base = qkv + (size_t)(b * T_SEQ) * rstride + h * HD;

  // ---- Q tile (64x64) via async copies: 4 x b128 per thread ----
  {
    const int r  = tid >> 1;
    const int c0 = (tid & 1) * 32;
    const bf16_t* src = base + (size_t)(q0 + r) * rstride + c0;
    async_b128(src,      &qs[r][c0]);
    async_b128(src + 8,  &qs[r][c0 + 8]);
    async_b128(src + 16, &qs[r][c0 + 16]);
    async_b128(src + 24, &qs[r][c0 + 24]);
  }
  if (tid < 64) { mrow[tid] = NEG_BIG; lrow[tid] = 0.0f; }

  v8f o[4] = {};
  async_wait0();
  __syncthreads();

  const int nkt = (q0 + 64) / 32;       // causal: keys 0 .. q0+63
  for (int kt = 0; kt < nkt; ++kt) {
    const int kbase = kt * 32;
    // ---- K async to LDS; V through VGPRs (needs transpose) ----
    {
      const int r  = tid >> 2;                  // key 0..31
      const int c0 = (tid & 3) * 16;            // d
      const bf16_t* krow = base + (size_t)(kbase + r) * rstride + CDIM + c0;
      async_b128(krow,     &ks[r][c0]);
      async_b128(krow + 8, &ks[r][c0 + 8]);
      const bf16_t* vrow = base + (size_t)(kbase + r) * rstride + 2 * CDIM + c0;
      v16bf vv = *(const v16bf*)vrow;
      #pragma unroll
      for (int e = 0; e < 16; ++e) vt[c0 + e][r] = vv[e];
    }
    async_wait0();
    __syncthreads();

    // ---- S = Q Kt : 16x32 per wave, hd=64 contraction (2 wmma steps) ----
    v8f sf0 = {}, sf1 = {};
    {
      const int qrow = 16 * wave + l16;
      #pragma unroll
      for (int s = 0; s < 2; ++s) {
        v8bf alo = *(const v8bf*)&qs[qrow][s * 32 + hi * 16];
        v8bf ahi = *(const v8bf*)&qs[qrow][s * 32 + hi * 16 + 8];
        v16bf af = join8(alo, ahi);
        v16bf b0 = *(const v16bf*)&ks[l16][s * 32 + hi * 16];
        v16bf b1 = *(const v16bf*)&ks[16 + l16][s * 32 + hi * 16];
        sf0 = __builtin_amdgcn_wmma_f32_16x16x32_bf16(false, af, false, b0,
                                                      (short)0, sf0, false, false);
        sf1 = __builtin_amdgcn_wmma_f32_16x16x32_bf16(false, af, false, b1,
                                                      (short)0, sf1, false, false);
      }
    }
    // ---- scale + emphasis(col 0) + causal mask -> LDS ----
    {
      const int rloc = 16 * wave + hi * 8;
      #pragma unroll
      for (int v = 0; v < 8; ++v) {
        const int rq = q0 + rloc + v;
        int ck = kbase + l16;
        float val = sf0[v] * 0.125f;
        if (ck == 0) val += 1.0f;               // EMPHASIS on key column 0
        if (ck > rq) val = NEG_BIG;             // causal
        sS[rloc + v][l16] = val;

        ck = kbase + 16 + l16;
        val = sf1[v] * 0.125f;
        if (ck == 0) val += 1.0f;
        if (ck > rq) val = NEG_BIG;
        sS[rloc + v][16 + l16] = val;
      }
    }
    __syncthreads();

    // ---- online softmax: one thread per query row ----
    if (tid < 64) {
      float mo = mrow[tid];
      float mn = mo;
      #pragma unroll
      for (int c = 0; c < 32; ++c) mn = fmaxf(mn, sS[tid][c]);
      const float al = __expf(mo - mn);
      float sum = 0.0f;
      #pragma unroll
      for (int c = 0; c < 32; ++c) {
        const float p = __expf(sS[tid][c] - mn);
        sum += p;
        pb[tid][c] = (bf16_t)p;
      }
      mrow[tid] = mn;
      lrow[tid] = lrow[tid] * al + sum;
      arow[tid] = al;
    }
    __syncthreads();

    // ---- rescale O by alpha, then O += P @ V ----
    {
      float av[8];
      #pragma unroll
      for (int v = 0; v < 8; ++v) av[v] = arow[16 * wave + hi * 8 + v];
      #pragma unroll
      for (int j = 0; j < 4; ++j)
        #pragma unroll
        for (int v = 0; v < 8; ++v) o[j][v] *= av[v];

      const int pr = 16 * wave + l16;
      v8bf plo = *(const v8bf*)&pb[pr][hi * 16];
      v8bf phi = *(const v8bf*)&pb[pr][hi * 16 + 8];
      v16bf pf = join8(plo, phi);
      #pragma unroll
      for (int j = 0; j < 4; ++j) {
        v16bf bfrag = *(const v16bf*)&vt[16 * j + l16][hi * 16];
        o[j] = __builtin_amdgcn_wmma_f32_16x16x32_bf16(
            false, pf, false, bfrag, (short)0, o[j], false, false);
      }
    }
    __syncthreads();
  }

  // ---- normalize by l, store bf16 ----
  {
    float linv[8];
    #pragma unroll
    for (int v = 0; v < 8; ++v) linv[v] = 1.0f / lrow[16 * wave + hi * 8 + v];
    #pragma unroll
    for (int j = 0; j < 4; ++j) {
      #pragma unroll
      for (int v = 0; v < 8; ++v) {
        const int trow = q0 + 16 * wave + hi * 8 + v;
        const int col  = h * HD + 16 * j + l16;
        out[(size_t)(b * T_SEQ + trow) * CDIM + col] = (bf16_t)(o[j][v] * linv[v]);
      }
    }
  }
}

// ---------------------------------------------------------------------------
// Workspace (48 MB):
//   [ 0,  8M)  x_bf       bf16 [4096,1024]
//   [ 8, 14M)  Wt_attn    bf16 [3072,1024]   (W_attn transposed)
//   [14, 16M)  Wt_proj    bf16 [1024,1024]   (W_proj transposed)
//   [16, 40M)  qkv        bf16 [4096,3072]
//   [40, 48M)  attn       bf16 [4096,1024]
// ---------------------------------------------------------------------------
extern "C" void kernel_launch(void* const* d_in, const int* in_sizes, int n_in,
                              void* d_out, int out_size, void* d_ws, size_t ws_size,
                              hipStream_t stream) {
  const float* x      = (const float*)d_in[0];   // [2,2048,1024]
  const float* W_attn = (const float*)d_in[1];   // [1024,3072]
  const float* W_proj = (const float*)d_in[2];   // [1024,1024]
  // d_in[3] = attn_mask (bool tril) -- causal mask computed analytically.
  float* out = (float*)d_out;

  char* ws = (char*)d_ws;
  bf16_t* xb  = (bf16_t*)(ws);
  bf16_t* wta = (bf16_t*)(ws + ((size_t)8  << 20));
  bf16_t* wtp = (bf16_t*)(ws + ((size_t)14 << 20));
  bf16_t* qkv = (bf16_t*)(ws + ((size_t)16 << 20));
  bf16_t* atb = (bf16_t*)(ws + ((size_t)40 << 20));

  // 0) one-time precision/layout conversion
  cvt_f32_to_bf16<<<dim3(BROWS * CDIM / 4 / 256), 256, 0, stream>>>(
      x, xb, BROWS * CDIM);
  transpose_cvt<<<dim3(3 * CDIM / 32, CDIM / 32), 256, 0, stream>>>(
      W_attn, wta, CDIM, 3 * CDIM);
  transpose_cvt<<<dim3(CDIM / 32, CDIM / 32), 256, 0, stream>>>(
      W_proj, wtp, CDIM, CDIM);

  // 1) qkv = x @ W_attn   (M=4096, N=3072, K=1024) -> bf16
  gemm_bf16_wmma<bf16_t><<<dim3(3 * CDIM / 128, BROWS / 128), 256, 0, stream>>>(
      xb, wta, qkv, BROWS, 3 * CDIM, CDIM);

  // 2) flash attention -> bf16
  attn_flash_wmma<<<dim3(T_SEQ / 64, NHEAD, 2), 128, 0, stream>>>(qkv, atb);

  // 3) y = attn @ W_proj  (M=4096, N=1024, K=1024) -> f32 output
  gemm_bf16_wmma<float><<<dim3(CDIM / 128, BROWS / 128), 256, 0, stream>>>(
      atb, wtp, out, BROWS, CDIM, CDIM);
}
